// TransformerBlock_62723702390802
// MI455X (gfx1250) — matrix operand order repaired and checked by reading, not compiled, not run on hardware
//
#include <hip/hip_runtime.h>

// ---------------------------------------------------------------------------
// Transformer block (ViT) for MI455X / gfx1250, bf16 WMMA everywhere.
// B=8, N=1024, C=768, H=12, Dh=64, HID=3072, M = B*N = 8192 tokens.
// GEMM tiles use CDNA5 async load-to-LDS (ASYNCcnt) with double buffering
// when the builtins are available; falls back to the proven sync path.
// ---------------------------------------------------------------------------

typedef __attribute__((ext_vector_type(16))) __bf16 v16bf;
typedef __attribute__((ext_vector_type(8)))  float  v8f;
typedef int i32x4 __attribute__((vector_size(16)));

union FragAB { unsigned u[8]; v16bf v; };

#define AS1 __attribute__((address_space(1)))
#define AS3 __attribute__((address_space(3)))

#if __has_builtin(__builtin_amdgcn_global_load_async_to_lds_b128)
#define ASYNC_TILE 1
#else
#define ASYNC_TILE 0
#endif

#if __has_builtin(__builtin_amdgcn_s_wait_asynccnt)
#define WAIT_ASYNC(n) __builtin_amdgcn_s_wait_asynccnt(n)
#else
#define WAIT_ASYNC(n) asm volatile("s_wait_asynccnt %0" ::"i"(n) : "memory")
#endif

__device__ __forceinline__ void async_cp16(const unsigned short* g, unsigned* l) {
#if ASYNC_TILE
    __builtin_amdgcn_global_load_async_to_lds_b128((AS1 i32x4*)g, (AS3 i32x4*)l,
                                                   0, 0);
#endif
}

__device__ __forceinline__ unsigned short f2bf(float f) {
    unsigned u = __builtin_bit_cast(unsigned, f);
    return (unsigned short)((u + 0x7FFFu + ((u >> 16) & 1u)) >> 16);
}

// ---------------------------------------------------------------------------
// fp32 -> bf16 elementwise convert (weights)
// ---------------------------------------------------------------------------
__global__ __launch_bounds__(256) void cvt_bf16(const float* __restrict__ src,
                                                unsigned short* __restrict__ dst,
                                                int n) {
    int i = blockIdx.x * 256 + threadIdx.x;
    if (i < n) dst[i] = f2bf(src[i]);
}

// ---------------------------------------------------------------------------
// LayerNorm over 768 features, one block per token, bf16 output.
// ---------------------------------------------------------------------------
__global__ __launch_bounds__(256) void ln_bf16(const float* __restrict__ x,
                                               const float* __restrict__ w,
                                               const float* __restrict__ b,
                                               unsigned short* __restrict__ y) {
    const int t = blockIdx.x;
    const float* xr = x + (size_t)t * 768;
    const int tid = threadIdx.x;
    float v0 = xr[tid], v1 = xr[tid + 256], v2 = xr[tid + 512];
    float s = v0 + v1 + v2;
    float s2 = v0 * v0 + v1 * v1 + v2 * v2;
#pragma unroll
    for (int off = 16; off > 0; off >>= 1) {
        s  += __shfl_xor(s,  off, 32);
        s2 += __shfl_xor(s2, off, 32);
    }
    __shared__ float red[8], red2[8];
    if ((tid & 31) == 0) { red[tid >> 5] = s; red2[tid >> 5] = s2; }
    __syncthreads();
    float S = 0.f, S2 = 0.f;
#pragma unroll
    for (int i = 0; i < 8; i++) { S += red[i]; S2 += red2[i]; }
    float mu  = S * (1.0f / 768.0f);
    float var = S2 * (1.0f / 768.0f) - mu * mu;
    float rs  = rsqrtf(var + 1e-5f);
    unsigned short* yr = y + (size_t)t * 768;
    yr[tid]       = f2bf((v0 - mu) * rs * w[tid]       + b[tid]);
    yr[tid + 256] = f2bf((v1 - mu) * rs * w[tid + 256] + b[tid + 256]);
    yr[tid + 512] = f2bf((v2 - mu) * rs * w[tid + 512] + b[tid + 512]);
}

// ---------------------------------------------------------------------------
// NT GEMM: C[M,Nc] = A[M,K](bf16) * W[Nc,K](bf16)^T, tiles 128x128x32.
// 256 threads = 8 waves in a 4x2 grid; each wave: 32x64 via 2x4 WMMA tiles.
// MODE 0: qkv scatter (q scaled, v transposed). MODE 1: +bias+resid -> fp32.
// MODE 2: +bias, exact GELU -> bf16.
// LDS tiles padded to 40 halves/row (20 words) -> conflict-free frag reads.
// Double-buffered async global->LDS when available.
// ---------------------------------------------------------------------------
#define TILE_W 2560  // 128 rows * 20 words

template <int MODE>
__global__ __launch_bounds__(256) void gemm_nt(
    const unsigned short* __restrict__ A, const unsigned short* __restrict__ Bw,
    int K, int ldc, const float* __restrict__ bias,
    const float* __restrict__ resid, float* __restrict__ outF,
    unsigned short* __restrict__ outBf, unsigned short* __restrict__ outQ,
    unsigned short* __restrict__ outKk, unsigned short* __restrict__ outV) {
    __shared__ unsigned shA[2 * TILE_W];
    __shared__ unsigned shB[2 * TILE_W];
    const int tid = threadIdx.x;
    const int lane = tid & 31, wave = tid >> 5;
    const int wm = wave >> 1, wn = wave & 1;
    const int l16 = lane & 15, lhi = lane >> 4;
    const int mBase = blockIdx.x * 128, nBase = blockIdx.y * 128;

    const v8f vz = {0.f, 0.f, 0.f, 0.f, 0.f, 0.f, 0.f, 0.f};
    v8f acc[2][4];
#pragma unroll
    for (int i = 0; i < 2; i++)
#pragma unroll
        for (int j = 0; j < 4; j++) acc[i][j] = vz;

    const int nk = K >> 5;

#if ASYNC_TILE
    auto issue = [&](int ki, int buf) {
        const int k0 = ki * 32;
#pragma unroll
        for (int i = 0; i < 2; i++) {
            int chunk = tid + i * 256;          // 512 x 16B chunks per tile
            int row = chunk >> 2, seg = chunk & 3;
            int wbase = buf * TILE_W + row * 20 + seg * 4;
            async_cp16(A + (size_t)(mBase + row) * K + k0 + seg * 8, &shA[wbase]);
            async_cp16(Bw + (size_t)(nBase + row) * K + k0 + seg * 8, &shB[wbase]);
        }
    };
    issue(0, 0);
#endif

    for (int ki = 0; ki < nk; ki++) {
        int bufw = 0;
#if ASYNC_TILE
        bufw = (ki & 1) * TILE_W;
        if (ki + 1 < nk) {
            issue(ki + 1, (ki + 1) & 1);  // prefetch next tile into other buffer
            WAIT_ASYNC(4);                // wait for current tile (in-order cnt)
        } else {
            WAIT_ASYNC(0);
        }
        __syncthreads();
#else
        const int k0 = ki * 32;
#pragma unroll
        for (int i = 0; i < 2; i++) {
            int chunk = tid + i * 256;
            int row = chunk >> 2, seg = chunk & 3;
            int wbase = row * 20 + seg * 4;
            uint4 va = *reinterpret_cast<const uint4*>(
                A + (size_t)(mBase + row) * K + k0 + seg * 8);
            shA[wbase + 0] = va.x; shA[wbase + 1] = va.y;
            shA[wbase + 2] = va.z; shA[wbase + 3] = va.w;
            uint4 vb = *reinterpret_cast<const uint4*>(
                Bw + (size_t)(nBase + row) * K + k0 + seg * 8);
            shB[wbase + 0] = vb.x; shB[wbase + 1] = vb.y;
            shB[wbase + 2] = vb.z; shB[wbase + 3] = vb.w;
        }
        __syncthreads();
#endif

        FragAB af[2], bfm[4];
#pragma unroll
        for (int ms = 0; ms < 2; ms++) {
            int r = wm * 32 + ms * 16 + l16;
            int base = bufw + r * 20 + lhi * 8; // kb = lhi*16 halves = lhi*8 words
#pragma unroll
            for (int j = 0; j < 8; j++) {
                int kw = j;  // words {0..7}
                af[ms].u[j] = shA[base + kw];
            }
        }
#pragma unroll
        for (int ns = 0; ns < 4; ns++) {
            int c = wn * 64 + ns * 16 + l16;
            int base = bufw + c * 20 + lhi * 8; // kb = lhi*16 halves = lhi*8 words
#pragma unroll
            for (int j = 0; j < 8; j++) bfm[ns].u[j] = shB[base + j];
        }
#pragma unroll
        for (int ms = 0; ms < 2; ms++)
#pragma unroll
            for (int ns = 0; ns < 4; ns++)
                acc[ms][ns] = __builtin_amdgcn_wmma_f32_16x16x32_bf16(
                    false, af[ms].v, false, bfm[ns].v, (short)0, acc[ms][ns],
                    false, false);
        __syncthreads();
    }

    // Epilogue. C layout: VGPR r -> row = r + 8*lhi, col = l16.
#pragma unroll
    for (int ms = 0; ms < 2; ms++)
#pragma unroll
        for (int ns = 0; ns < 4; ns++)
#pragma unroll
            for (int r = 0; r < 8; r++) {
                int row = mBase + wm * 32 + ms * 16 + lhi * 8 + r;
                int col = nBase + wn * 64 + ns * 16 + l16;
                float val = acc[ms][ns][r] + bias[col];
                if (MODE == 0) {
                    int bb = row >> 10, n = row & 1023;
                    if (col < 768) {
                        int h = col >> 6, d = col & 63;
                        outQ[(((size_t)(bb * 12 + h)) * 1024 + n) * 64 + d] =
                            f2bf(val * 0.125f);  // Dh^-0.5 folded into Q
                    } else if (col < 1536) {
                        int c2 = col - 768; int h = c2 >> 6, d = c2 & 63;
                        outKk[(((size_t)(bb * 12 + h)) * 1024 + n) * 64 + d] =
                            f2bf(val);
                    } else {
                        int c2 = col - 1536; int h = c2 >> 6, d = c2 & 63;
                        outV[(((size_t)(bb * 12 + h)) * 64 + d) * 1024 + n] =
                            f2bf(val);           // V stored transposed
                    }
                } else if (MODE == 1) {
                    size_t idx = (size_t)row * ldc + col;
                    outF[idx] = val + resid[idx];
                } else {
                    float g = 0.5f * val * (1.0f + erff(val * 0.70710678118f));
                    outBf[(size_t)row * ldc + col] = f2bf(g);
                }
            }
}

// ---------------------------------------------------------------------------
// Flash attention: grid = B*H*(N/128); block = 256 = 8 waves, each wave owns
// 16 query rows. q,k: [bh][n][64] bf16; vt: [bh][64][n] bf16 (transposed).
// Output attn: [b*n][h*64+d] bf16 for the proj GEMM.
// ---------------------------------------------------------------------------
__global__ __launch_bounds__(256) void attn_kernel(
    const unsigned short* __restrict__ q, const unsigned short* __restrict__ kk,
    const unsigned short* __restrict__ vt, unsigned short* __restrict__ attn) {
    __shared__ unsigned pbuf[8][16 * 36];  // per-wave P slice, 16 x 72 halves
    const int tid = threadIdx.x, lane = tid & 31, wave = tid >> 5;
    const int l16 = lane & 15, lhi = lane >> 4;
    const int mblk = blockIdx.x & 7, bh = blockIdx.x >> 3;
    const size_t hb = (size_t)bh * 1024 * 64;
    const int qrow0 = mblk * 128 + wave * 16;

    // Q A-fragments for both 32-wide K steps of d=64, loaded once.
    FragAB aq[2];
    {
        const unsigned* qp = reinterpret_cast<const unsigned*>(
            q + hb + (size_t)(qrow0 + l16) * 64);
#pragma unroll
        for (int st = 0; st < 2; st++)
#pragma unroll
            for (int j = 0; j < 8; j++) {
                int ks = lhi * 16 + 2 * j + st * 32;
                aq[st].u[j] = qp[ks >> 1];
            }
    }

    const v8f vz = {0.f, 0.f, 0.f, 0.f, 0.f, 0.f, 0.f, 0.f};
    v8f o[4];
    float mrow[8], lrow[8];
#pragma unroll
    for (int dt = 0; dt < 4; dt++) o[dt] = vz;
#pragma unroll
    for (int r = 0; r < 8; r++) { mrow[r] = -1e30f; lrow[r] = 0.f; }

    unsigned* pw = &pbuf[wave][0];

    for (int kt = 0; kt < 16; kt++) {       // 16 key tiles of 64
        v8f s[4];
#pragma unroll
        for (int nt = 0; nt < 4; nt++) s[nt] = vz;
#pragma unroll
        for (int st = 0; st < 2; st++) {
            FragAB bk[4];
#pragma unroll
            for (int nt = 0; nt < 4; nt++) {
                const unsigned* kp = reinterpret_cast<const unsigned*>(
                    kk + hb + (size_t)(kt * 64 + nt * 16 + l16) * 64);
#pragma unroll
                for (int j = 0; j < 8; j++) {
                    int ks = lhi * 16 + 2 * j + st * 32;
                    bk[nt].u[j] = kp[ks >> 1];
                }
            }
#pragma unroll
            for (int nt = 0; nt < 4; nt++)
                s[nt] = __builtin_amdgcn_wmma_f32_16x16x32_bf16(
                    false, aq[st].v, false, bk[nt].v, (short)0, s[nt], false,
                    false);
        }
        // ---- online softmax (rows exclusive to this wave's 16-lane groups)
        float scl[8], ps[8];
#pragma unroll
        for (int r = 0; r < 8; r++) {
            float m4 = fmaxf(fmaxf(s[0][r], s[1][r]), fmaxf(s[2][r], s[3][r]));
#pragma unroll
            for (int off = 8; off > 0; off >>= 1)
                m4 = fmaxf(m4, __shfl_xor(m4, off, 32));
            float mn = fmaxf(mrow[r], m4);
            scl[r] = __expf(mrow[r] - mn);
            mrow[r] = mn;
            ps[r] = 0.f;
        }
        // P = exp(S - m); pack lane pairs into 32-bit LDS words (A-frag layout)
#pragma unroll
        for (int nt = 0; nt < 4; nt++)
#pragma unroll
            for (int r = 0; r < 8; r++) {
                float p = __expf(s[nt][r] - mrow[r]);
                ps[r] += p;
                unsigned pb = (unsigned)f2bf(p);
                unsigned partner = __shfl_xor(pb, 1, 32);
                unsigned packed = (l16 & 1) ? ((partner & 0xFFFFu) | (pb << 16))
                                            : (pb | (partner << 16));
                int rowi = lhi * 8 + r;
                int colw = (nt * 16 + l16) >> 1;
                pw[rowi * 36 + colw] = packed;
            }
#pragma unroll
        for (int r = 0; r < 8; r++) {
            float t = ps[r];
#pragma unroll
            for (int off = 8; off > 0; off >>= 1) t += __shfl_xor(t, off, 32);
            lrow[r] = lrow[r] * scl[r] + t;
        }
#pragma unroll
        for (int dt = 0; dt < 4; dt++)
#pragma unroll
            for (int r = 0; r < 8; r++) o[dt][r] *= scl[r];

        asm volatile("s_wait_dscnt 0" ::: "memory");  // P stores visible

        // ---- O += P * V  (contraction over 64 keys, two 32-wide steps)
#pragma unroll
        for (int st = 0; st < 2; st++) {
            FragAB ap;
#pragma unroll
            for (int j = 0; j < 8; j++) {
                int ks = lhi * 16 + 2 * j + st * 32;
                ap.u[j] = pw[l16 * 36 + (ks >> 1)];
            }
#pragma unroll
            for (int dt = 0; dt < 4; dt++) {
                FragAB bv;
                const unsigned* vp = reinterpret_cast<const unsigned*>(
                    vt + hb + (size_t)(dt * 16 + l16) * 1024);
#pragma unroll
                for (int j = 0; j < 8; j++) {
                    int ks = kt * 64 + st * 32 + lhi * 16 + 2 * j;
                    bv.u[j] = vp[ks >> 1];
                }
                o[dt] = __builtin_amdgcn_wmma_f32_16x16x32_bf16(
                    false, ap.v, false, bv.v, (short)0, o[dt], false, false);
            }
        }
    }

    const int bb = bh / 12, h = bh % 12;
#pragma unroll
    for (int dt = 0; dt < 4; dt++)
#pragma unroll
        for (int r = 0; r < 8; r++) {
            int n = qrow0 + lhi * 8 + r;
            size_t t = (size_t)bb * 1024 + n;
            attn[t * 768 + h * 64 + dt * 16 + l16] = f2bf(o[dt][r] / lrow[r]);
        }
}

// ---------------------------------------------------------------------------
extern "C" void kernel_launch(void* const* d_in, const int* in_sizes, int n_in,
                              void* d_out, int out_size, void* d_ws,
                              size_t ws_size, hipStream_t stream) {
    (void)in_sizes; (void)n_in; (void)out_size; (void)ws_size;
    const float* x     = (const float*)d_in[0];
    const float* ln1w  = (const float*)d_in[1];
    const float* ln1b  = (const float*)d_in[2];
    const float* qkvw  = (const float*)d_in[3];
    const float* qkvb  = (const float*)d_in[4];
    const float* projw = (const float*)d_in[5];
    const float* projb = (const float*)d_in[6];
    const float* ln2w  = (const float*)d_in[7];
    const float* ln2b  = (const float*)d_in[8];
    const float* fc1w  = (const float*)d_in[9];
    const float* fc1b  = (const float*)d_in[10];
    const float* fc2w  = (const float*)d_in[11];
    const float* fc2b  = (const float*)d_in[12];

    char* ws = (char*)d_ws;
    size_t off = 0;
    auto take = [&](size_t bytes) -> char* {
        char* p = ws + off;
        off += (bytes + 255) & ~(size_t)255;
        return p;
    };
    unsigned short* wQ   = (unsigned short*)take((size_t)2304 * 768 * 2);
    unsigned short* wP   = (unsigned short*)take((size_t)768 * 768 * 2);
    unsigned short* w1   = (unsigned short*)take((size_t)3072 * 768 * 2);
    unsigned short* w2   = (unsigned short*)take((size_t)768 * 3072 * 2);
    unsigned short* xln  = (unsigned short*)take((size_t)8192 * 768 * 2);
    unsigned short* qb   = (unsigned short*)take((size_t)8192 * 768 * 2);
    unsigned short* kb   = (unsigned short*)take((size_t)8192 * 768 * 2);
    unsigned short* vtb  = (unsigned short*)take((size_t)8192 * 768 * 2);
    unsigned short* attnb= (unsigned short*)take((size_t)8192 * 768 * 2);
    float*          x1   = (float*)take((size_t)8192 * 768 * 4);
    unsigned short* h1   = qb;  // reuse q|k|vt|attn region (exactly 8192*3072*2)

    // 1) weights -> bf16
    cvt_bf16<<<(2304 * 768 + 255) / 256, 256, 0, stream>>>(qkvw, wQ, 2304 * 768);
    cvt_bf16<<<(768 * 768 + 255) / 256, 256, 0, stream>>>(projw, wP, 768 * 768);
    cvt_bf16<<<(3072 * 768 + 255) / 256, 256, 0, stream>>>(fc1w, w1, 3072 * 768);
    cvt_bf16<<<(768 * 3072 + 255) / 256, 256, 0, stream>>>(fc2w, w2, 768 * 3072);
    // 2) LN1
    ln_bf16<<<8192, 256, 0, stream>>>(x, ln1w, ln1b, xln);
    // 3) QKV GEMM (scatter q/k/vT)
    gemm_nt<0><<<dim3(64, 18), 256, 0, stream>>>(xln, wQ, 768, 0, qkvb, nullptr,
                                                 nullptr, nullptr, qb, kb, vtb);
    // 4) attention
    attn_kernel<<<8 * 12 * 8, 256, 0, stream>>>(qb, kb, vtb, attnb);
    // 5) proj + residual -> x1 (fp32)
    gemm_nt<1><<<dim3(64, 6), 256, 0, stream>>>(attnb, wP, 768, 768, projb, x,
                                                x1, nullptr, nullptr, nullptr,
                                                nullptr);
    // 6) LN2 (reuse xln buffer)
    ln_bf16<<<8192, 256, 0, stream>>>(x1, ln2w, ln2b, xln);
    // 7) fc1 + GELU -> h1 (bf16)
    gemm_nt<2><<<dim3(64, 24), 256, 0, stream>>>(xln, w1, 768, 3072, fc1b,
                                                 nullptr, nullptr, h1, nullptr,
                                                 nullptr, nullptr);
    // 8) fc2 + residual -> d_out (fp32)
    gemm_nt<1><<<dim3(64, 6), 256, 0, stream>>>(h1, w2, 3072, 768, fc2b, x1,
                                                (float*)d_out, nullptr, nullptr,
                                                nullptr, nullptr);
}
